// GPSA_51556787421954
// MI455X (gfx1250) — compile-verified, matrix operand order checked by reading, not run on hardware
//
#include <hip/hip_runtime.h>

#define B_  4
#define N_  1024
#define C_  256
#define H_  8
#define HD_ 32
#define SCALE_ 0.17677669529663687f  /* 1/sqrt(32) */

typedef __attribute__((ext_vector_type(16))) __bf16 v16bf;
typedef __attribute__((ext_vector_type(8)))  float  v8f;
typedef __attribute__((ext_vector_type(4)))  int    v4i;

union BF16x16 { v16bf v; uint4 q[2]; __bf16 h[16]; };

#if defined(__has_builtin)
#if __has_builtin(__builtin_amdgcn_global_load_async_to_lds_b128)
#define HAVE_ASYNC 1
#endif
#endif

#ifdef HAVE_ASYNC
__device__ inline void async_copy16(const void* g, void* l) {
    __builtin_amdgcn_global_load_async_to_lds_b128(
        (__attribute__((address_space(1))) v4i*)(void*)g,
        (__attribute__((address_space(3))) v4i*)l, 0, 0);
}
#endif

__device__ inline float rmax16(float v) {
#pragma unroll
    for (int m = 1; m < 16; m <<= 1) v = fmaxf(v, __shfl_xor(v, m, 32));
    return v;
}
__device__ inline float rsum16(float v) {
#pragma unroll
    for (int m = 1; m < 16; m <<= 1) v += __shfl_xor(v, m, 32);
    return v;
}

// ---------------------------------------------------------------------------
// Prep kernels: one-time f32->bf16 conversion of x, and convert+transpose of
// the weight matrices so GEMM B-operands are contiguous 16B runs per lane.
// ---------------------------------------------------------------------------
__global__ void cvt_x_kernel(const float* __restrict__ x, __bf16* __restrict__ xb)
{
    int idx = (blockIdx.x * blockDim.x + threadIdx.x) * 4;   // over 1M elems
    float4 f = *(const float4*)(x + idx);
    union { __bf16 h[4]; unsigned long long u; } o;
    o.h[0] = (__bf16)f.x; o.h[1] = (__bf16)f.y;
    o.h[2] = (__bf16)f.z; o.h[3] = (__bf16)f.w;
    *(unsigned long long*)(xb + idx) = o.u;
}

// dst[c][k] = src[k][c], dst bf16, src is [256 x cols] f32 row-major
__global__ void wtrans_kernel(const float* __restrict__ w, __bf16* __restrict__ wt,
                              int cols)
{
    int idx = blockIdx.x * blockDim.x + threadIdx.x;
    if (idx >= 256 * cols) return;
    int k = idx & 255;
    int c = idx >> 8;
    wt[idx] = (__bf16)w[(size_t)k * cols + c];
}

// A[b,h,n] = sum_c vox[b,n,c]*W_pos[c,h]  (separable part of pos logits)
__global__ void pos_pre_kernel(const float* __restrict__ vox,
                               const float* __restrict__ Wpos,
                               float* __restrict__ apos)
{
    int idx = blockIdx.x * blockDim.x + threadIdx.x;    // b*H*N + h*N + n
    if (idx >= B_ * H_ * N_) return;
    int n  = idx & (N_ - 1);
    int bh = idx >> 10;
    int h  = bh & (H_ - 1);
    int b  = bh >> 3;
    const float* p = vox + ((size_t)b * N_ + n) * 3;
    apos[idx] = p[0] * Wpos[0 * H_ + h] + p[1] * Wpos[1 * H_ + h] + p[2] * Wpos[2 * H_ + h];
}

// ---------------------------------------------------------------------------
// Kernel: QKV projection. xb[4096x256]bf16 @ [Wqkt|Wvt] -> Q,K [B,H,N,HD],
// V d-major [B,H,HD,N]. One wave per 16x16 tile; 4 b128 loads + 1 wmma / chunk.
// ---------------------------------------------------------------------------
__global__ __launch_bounds__(128) void qkv_proj_kernel(
    const __bf16* __restrict__ xb, const __bf16* __restrict__ Wqkt,
    const __bf16* __restrict__ Wvt,
    __bf16* __restrict__ qg, __bf16* __restrict__ kg, __bf16* __restrict__ vg)
{
    const int lane = threadIdx.x & 31;
    const int wid  = blockIdx.x * 4 + (threadIdx.x >> 5);
    const int it = wid / 48;          // 256 row tiles
    const int nt = wid % 48;          // 48 col tiles (768 cols)
    const int ibase = it * 16;
    const int nbase = nt * 16;
    const int ln = lane & 15;
    const int hi = lane >> 4;
    const int kb = (lane < 16) ? 0 : 8;
    const int col = nbase + ln;

    const __bf16* ap = xb + (size_t)(ibase + ln) * C_;
    const __bf16* bp = (nbase < 512) ? (Wqkt + (size_t)col * C_)
                                     : (Wvt + (size_t)(col - 512) * C_);
    v8f acc = {};
    for (int k = 0; k < C_; k += 32) {
        BF16x16 a, bb;
        a.q[0]  = *(const uint4*)(ap + k + kb);
        a.q[1]  = *(const uint4*)(ap + k + kb + 16);
        bb.q[0] = *(const uint4*)(bp + k + kb);
        bb.q[1] = *(const uint4*)(bp + k + kb + 16);
        acc = __builtin_amdgcn_wmma_f32_16x16x32_bf16(false, a.v, false, bb.v,
                                                      (short)0, acc, false, false);
    }
#pragma unroll
    for (int r = 0; r < 8; ++r) {
        int i = ibase + r + 8 * hi;         // 0..4095 = b*1024 + n
        int b = i >> 10;
        int n = i & 1023;
        __bf16 val = (__bf16)acc[r];
        if (col < 256) {
            int h = col >> 5, d = col & 31;
            qg[(((size_t)b * H_ + h) * N_ + n) * HD_ + d] = val;
        } else if (col < 512) {
            int cc = col - 256, h = cc >> 5, d = cc & 31;
            kg[(((size_t)b * H_ + h) * N_ + n) * HD_ + d] = val;
        } else {
            int cc = col - 512, h = cc >> 5, d = cc & 31;
            vg[(((size_t)b * H_ + h) * HD_ + d) * N_ + n] = val;
        }
    }
}

// ---------------------------------------------------------------------------
// Fused dual-softmax flash attention. One wave per (b,h,16-row query tile).
// out = (1-sig)*softmax(QK^T*s)V + sig*softmax(pos)V   (renormalizer == 1)
// K/V tiles staged to LDS via async-to-LDS when available.
// ---------------------------------------------------------------------------
__global__ __launch_bounds__(128) void attn_kernel(
    const __bf16* __restrict__ qg, const __bf16* __restrict__ kg,
    const __bf16* __restrict__ vg,
    const float* __restrict__ vox, const float* __restrict__ apos,
    const float* __restrict__ Wpos, const float* __restrict__ bpos,
    const float* __restrict__ gating, __bf16* __restrict__ aout)
{
    __shared__ __align__(64) __bf16 Plds[4][2][16 * 32];
#ifdef HAVE_ASYNC
    __shared__ __align__(64) __bf16 Klds[4][32 * 32];
    __shared__ __align__(64) __bf16 Vlds[4][32 * 32];
#endif

    const int lane  = threadIdx.x & 31;
    const int wslot = threadIdx.x >> 5;
    const int wid   = blockIdx.x * 4 + wslot;     // 0..2047
    const int it = wid & 63;
    const int h  = (wid >> 6) & 7;
    const int b  = wid >> 9;
    const int ibase = it * 16;
    const int ln = lane & 15;
    const int hi = lane >> 4;
    const int kb = (lane < 16) ? 0 : 8;
    const int bh = b * H_ + h;

    BF16x16 qa;
    {
        const __bf16* qp = qg + ((size_t)bh * N_ + (ibase + ln)) * HD_;
        qa.q[0] = *(const uint4*)(qp + kb);
        qa.q[1] = *(const uint4*)(qp + kb + 16);
    }
    float pix[8], piy[8], piz[8], Ai[8];
#pragma unroll
    for (int r = 0; r < 8; ++r) {
        int i = ibase + r + 8 * hi;
        const float* p = vox + ((size_t)b * N_ + i) * 3;
        pix[r] = p[0]; piy[r] = p[1]; piz[r] = p[2];
        Ai[r] = apos[(size_t)bh * N_ + i];
    }
    const float w3h = Wpos[3 * H_ + h];
    const float bph = bpos[h];

    float m1[8], l1[8], m2[8], l2[8];
    v8f a1a = {}, a1b = {}, a2a = {}, a2b = {};
#pragma unroll
    for (int r = 0; r < 8; ++r) { m1[r] = -1e30f; l1[r] = 0.f; m2[r] = -1e30f; l2[r] = 0.f; }

    for (int jc = 0; jc < N_; jc += 32) {
        BF16x16 k0, k1, v0, v1;
#ifdef HAVE_ASYNC
        // stage K tile [32j x 32d] and V tile [32d x 32j] into LDS via the
        // async-to-LDS engine; DScnt/ASYNCcnt hazards handled explicitly.
        asm volatile("s_wait_dscnt 0x0" ::: "memory");
        {
            const __bf16* krow = kg + ((size_t)bh * N_ + jc + lane) * HD_;
            const __bf16* vrow = vg + ((size_t)bh * HD_ + lane) * N_ + jc;
            __bf16* kl = &Klds[wslot][lane * 32];
            __bf16* vl = &Vlds[wslot][lane * 32];
#pragma unroll
            for (int qd = 0; qd < 4; ++qd) {
                async_copy16(krow + qd * 8, kl + qd * 8);
                async_copy16(vrow + qd * 8, vl + qd * 8);
            }
        }
        asm volatile("s_wait_asynccnt 0x0" ::: "memory");
        k0.q[0] = *(const uint4*)&Klds[wslot][ln * 32 + kb];
        k0.q[1] = *(const uint4*)&Klds[wslot][ln * 32 + kb + 16];
        k1.q[0] = *(const uint4*)&Klds[wslot][(16 + ln) * 32 + kb];
        k1.q[1] = *(const uint4*)&Klds[wslot][(16 + ln) * 32 + kb + 16];
        v0.q[0] = *(const uint4*)&Vlds[wslot][ln * 32 + kb];
        v0.q[1] = *(const uint4*)&Vlds[wslot][ln * 32 + kb + 16];
        v1.q[0] = *(const uint4*)&Vlds[wslot][(16 + ln) * 32 + kb];
        v1.q[1] = *(const uint4*)&Vlds[wslot][(16 + ln) * 32 + kb + 16];
#else
        {
            const __bf16* kp0 = kg + ((size_t)bh * N_ + (jc + ln)) * HD_;
            k0.q[0] = *(const uint4*)(kp0 + kb);
            k0.q[1] = *(const uint4*)(kp0 + kb + 16);
            const __bf16* kp1 = kg + ((size_t)bh * N_ + (jc + 16 + ln)) * HD_;
            k1.q[0] = *(const uint4*)(kp1 + kb);
            k1.q[1] = *(const uint4*)(kp1 + kb + 16);
            const __bf16* vp0 = vg + ((size_t)bh * HD_ + ln) * N_ + jc;
            v0.q[0] = *(const uint4*)(vp0 + kb);
            v0.q[1] = *(const uint4*)(vp0 + kb + 16);
            const __bf16* vp1 = vg + ((size_t)bh * HD_ + 16 + ln) * N_ + jc;
            v1.q[0] = *(const uint4*)(vp1 + kb);
            v1.q[1] = *(const uint4*)(vp1 + kb + 16);
        }
        if (jc + 32 < N_)
            __builtin_prefetch(kg + ((size_t)bh * N_ + (jc + 32 + ln)) * HD_, 0, 1);
#endif
        v8f z = {};
        v8f s0 = __builtin_amdgcn_wmma_f32_16x16x32_bf16(false, qa.v, false, k0.v,
                                                         (short)0, z, false, false);
        v8f s1 = __builtin_amdgcn_wmma_f32_16x16x32_bf16(false, qa.v, false, k1.v,
                                                         (short)0, z, false, false);

        // positional logits, directly in C layout
        float t0[8], t1[8];
        {
            const float* pj = vox + ((size_t)b * N_ + (jc + ln)) * 3;
            float jx = pj[0], jy = pj[1], jz = pj[2];
            float Aj = apos[(size_t)bh * N_ + (jc + ln)];
#pragma unroll
            for (int r = 0; r < 8; ++r) {
                float dx = jx - pix[r], dy = jy - piy[r], dz = jz - piz[r];
                t0[r] = Aj - Ai[r] + w3h * sqrtf(dx * dx + dy * dy + dz * dz) + bph;
            }
            const float* pj1 = vox + ((size_t)b * N_ + (jc + 16 + ln)) * 3;
            float jx1 = pj1[0], jy1 = pj1[1], jz1 = pj1[2];
            float Aj1 = apos[(size_t)bh * N_ + (jc + 16 + ln)];
#pragma unroll
            for (int r = 0; r < 8; ++r) {
                float dx = jx1 - pix[r], dy = jy1 - piy[r], dz = jz1 - piz[r];
                t1[r] = Aj1 - Ai[r] + w3h * sqrtf(dx * dx + dy * dy + dz * dz) + bph;
            }
        }

        // dual online softmax; P tiles -> LDS (bf16) for C->A relayout
#pragma unroll
        for (int r = 0; r < 8; ++r) {
            int rowoff = (r + 8 * hi) * 32 + ln;
            float sa = s0[r] * SCALE_, sb = s1[r] * SCALE_;
            float mn = fmaxf(m1[r], rmax16(fmaxf(sa, sb)));
            float p0 = __expf(sa - mn), p1 = __expf(sb - mn);
            float rs = rsum16(p0 + p1);
            float sc = __expf(m1[r] - mn);
            l1[r] = l1[r] * sc + rs;  m1[r] = mn;
            a1a[r] *= sc;  a1b[r] *= sc;
            Plds[wslot][0][rowoff]      = (__bf16)p0;
            Plds[wslot][0][rowoff + 16] = (__bf16)p1;

            float mn2 = fmaxf(m2[r], rmax16(fmaxf(t0[r], t1[r])));
            float q0 = __expf(t0[r] - mn2), q1 = __expf(t1[r] - mn2);
            float rs2 = rsum16(q0 + q1);
            float sc2 = __expf(m2[r] - mn2);
            l2[r] = l2[r] * sc2 + rs2;  m2[r] = mn2;
            a2a[r] *= sc2;  a2b[r] *= sc2;
            Plds[wslot][1][rowoff]      = (__bf16)q0;
            Plds[wslot][1][rowoff + 16] = (__bf16)q1;
        }
        __syncthreads();

        BF16x16 pa1, pa2;
        {
            const __bf16* pl1 = &Plds[wslot][0][ln * 32];
            pa1.q[0] = *(const uint4*)(pl1 + kb);
            pa1.q[1] = *(const uint4*)(pl1 + kb + 16);
            const __bf16* pl2 = &Plds[wslot][1][ln * 32];
            pa2.q[0] = *(const uint4*)(pl2 + kb);
            pa2.q[1] = *(const uint4*)(pl2 + kb + 16);
        }
        a1a = __builtin_amdgcn_wmma_f32_16x16x32_bf16(false, pa1.v, false, v0.v,
                                                      (short)0, a1a, false, false);
        a1b = __builtin_amdgcn_wmma_f32_16x16x32_bf16(false, pa1.v, false, v1.v,
                                                      (short)0, a1b, false, false);
        a2a = __builtin_amdgcn_wmma_f32_16x16x32_bf16(false, pa2.v, false, v0.v,
                                                      (short)0, a2a, false, false);
        a2b = __builtin_amdgcn_wmma_f32_16x16x32_bf16(false, pa2.v, false, v1.v,
                                                      (short)0, a2b, false, false);
        __syncthreads();
    }

    float g   = gating[h];
    float sig = 1.f / (1.f + __expf(-g));
    float g1 = 1.f - sig, g2 = sig;
#pragma unroll
    for (int r = 0; r < 8; ++r) {
        float inv1 = 1.f / l1[r], inv2 = 1.f / l2[r];
        int i = ibase + r + 8 * hi;
        size_t base = ((size_t)b * N_ + i) * C_ + h * HD_;
        aout[base + ln]      = (__bf16)(g1 * a1a[r] * inv1 + g2 * a2a[r] * inv2);
        aout[base + 16 + ln] = (__bf16)(g1 * a1b[r] * inv1 + g2 * a2b[r] * inv2);
    }
}

// ---------------------------------------------------------------------------
// Output projection: [4096x256]bf16 @ Wpt^T + b_proj -> f32 out.
// Bias pre-loaded into the WMMA C operand.
// ---------------------------------------------------------------------------
__global__ __launch_bounds__(128) void proj_kernel(
    const __bf16* __restrict__ aout, const __bf16* __restrict__ Wpt,
    const float* __restrict__ bp, float* __restrict__ out)
{
    const int lane = threadIdx.x & 31;
    const int wid  = blockIdx.x * 4 + (threadIdx.x >> 5);  // 0..4095
    const int it = wid >> 4;
    const int nt = wid & 15;
    const int ibase = it * 16, nbase = nt * 16;
    const int ln = lane & 15, hi = lane >> 4;
    const int kb = (lane < 16) ? 0 : 8;
    const int col = nbase + ln;

    float bias = bp[col];
    v8f acc;
#pragma unroll
    for (int r = 0; r < 8; ++r) acc[r] = bias;

    const __bf16* ap = aout + (size_t)(ibase + ln) * C_;
    const __bf16* wp = Wpt + (size_t)col * C_;
    for (int k = 0; k < C_; k += 32) {
        BF16x16 a, bb;
        a.q[0]  = *(const uint4*)(ap + k + kb);
        a.q[1]  = *(const uint4*)(ap + k + kb + 16);
        bb.q[0] = *(const uint4*)(wp + k + kb);
        bb.q[1] = *(const uint4*)(wp + k + kb + 16);
        acc = __builtin_amdgcn_wmma_f32_16x16x32_bf16(false, a.v, false, bb.v,
                                                      (short)0, acc, false, false);
    }
#pragma unroll
    for (int r = 0; r < 8; ++r)
        out[(size_t)(ibase + r + 8 * hi) * C_ + col] = acc[r];
}

// ---------------------------------------------------------------------------
extern "C" void kernel_launch(void* const* d_in, const int* in_sizes, int n_in,
                              void* d_out, int out_size, void* d_ws, size_t ws_size,
                              hipStream_t stream)
{
    const float* x     = (const float*)d_in[0];
    const float* vox   = (const float*)d_in[1];
    const float* Wqk   = (const float*)d_in[2];
    const float* Wv    = (const float*)d_in[3];
    const float* Wproj = (const float*)d_in[4];
    const float* bproj = (const float*)d_in[5];
    const float* Wpos  = (const float*)d_in[6];
    const float* bpos  = (const float*)d_in[7];
    const float* gate  = (const float*)d_in[8];
    float* out = (float*)d_out;

    char* w = (char*)d_ws;
    __bf16* xb   = (__bf16*)(w);                               // 2 MB
    __bf16* qg   = (__bf16*)(w + (2u << 20));                  // 2 MB
    __bf16* kg   = (__bf16*)(w + (4u << 20));                  // 2 MB
    __bf16* vg   = (__bf16*)(w + (6u << 20));                  // 2 MB
    __bf16* aob  = (__bf16*)(w + (8u << 20));                  // 2 MB
    __bf16* Wqkt = (__bf16*)(w + (10u << 20));                 // 256 KB
    __bf16* Wvt  = (__bf16*)(w + (10u << 20) + (256u << 10));  // 128 KB
    __bf16* Wpt  = (__bf16*)(w + (10u << 20) + (384u << 10));  // 128 KB
    float*  apos = (float*)(w + (10u << 20) + (512u << 10));   // 128 KB

    cvt_x_kernel<<<(B_ * N_ * C_ / 4 + 255) / 256, 256, 0, stream>>>(x, xb);
    wtrans_kernel<<<(256 * 512 + 255) / 256, 256, 0, stream>>>(Wqk, Wqkt, 512);
    wtrans_kernel<<<(256 * 256 + 255) / 256, 256, 0, stream>>>(Wv, Wvt, 256);
    wtrans_kernel<<<(256 * 256 + 255) / 256, 256, 0, stream>>>(Wproj, Wpt, 256);
    pos_pre_kernel<<<(B_ * H_ * N_ + 255) / 256, 256, 0, stream>>>(vox, Wpos, apos);

    qkv_proj_kernel<<<3072, 128, 0, stream>>>(xb, Wqkt, Wvt, qg, kg, vg);
    attn_kernel<<<512, 128, 0, stream>>>(qg, kg, vg, vox, apos, Wpos, bpos, gate, aob);
    proj_kernel<<<1024, 128, 0, stream>>>(aob, Wpt, bproj, out);
}
